// MultiheadAttention_40261023432752
// MI455X (gfx1250) — matrix operand order repaired and checked by reading, not compiled, not run on hardware
//
#include <hip/hip_runtime.h>
#include <math.h>

typedef __attribute__((ext_vector_type(16))) _Float16 v16h;
typedef __attribute__((ext_vector_type(8)))  _Float16 v8h;
typedef __attribute__((ext_vector_type(8)))  float    v8f;
typedef int v4i __attribute__((vector_size(16)));

#define B_SZ 8
#define NH   12
#define SEQ  1024
#define DH   64
#define CIN  768

#if defined(__gfx1250__)
#  if __has_builtin(__builtin_amdgcn_global_load_async_to_lds_b128) && \
      __has_builtin(__builtin_amdgcn_s_wait_asynccnt)
#    define USE_ASYNC 1
#  endif
#endif
#ifndef USE_ASYNC
#  define USE_ASYNC 0
#endif

// ISA 7.12.2: 16-bit A-matrix 16x32, lane L: M = L%16, half = L/16
//   element e of v16h: K = (e<8 ? e : e+8) + half*8   (two contiguous 8-runs)
__device__ __forceinline__ int a_kidx(int e, int lh) {
  return ((e < 8) ? e : (e + 8)) + (lh << 3);
}
// ISA 7.12.2: 16-bit B-matrix 32x16, lane L: N = L%16, half = L/16
//   element e: K = (e<8 ? e : e+8) + half*8   (same order as A)
__device__ __forceinline__ int b_kidx(int e, int lh) {
  return ((e < 8) ? e : (e + 8)) + (lh << 3);
}

__device__ __forceinline__ v8f wmma16(v16h a, v16h b, v8f c) {
  return __builtin_amdgcn_wmma_f32_16x16x32_f16(false, a, false, b, (short)0, c,
                                                false, false);
}

// ---------------------------------------------------------------------------
// Kernel 1: per-head projection.  out[b,h,s,d] = sum_c X[b,c,s]*W[h,c,d]+bias
// X is [B][C][S] (channel-major), so LDS staging performs the s<->c transpose.
// Weight tile staged TRANSPOSED (lBt[d][c]) so B-fragments are contiguous.
// Block: 256 threads (8 waves), tile 128(s) x 64(d), K-chunk 32.
// ---------------------------------------------------------------------------
__global__ __launch_bounds__(256)
void proj_kernel(const float* __restrict__ X, const float* __restrict__ W,
                 const float* __restrict__ bias, _Float16* __restrict__ out) {
  __shared__ alignas(16) _Float16 lA[128][40]; // [s][c], 80B row (16B mult)
  __shared__ alignas(16) _Float16 lBt[64][40]; // [d][c], transposed
  const int tid  = threadIdx.x;
  const int b    = blockIdx.z, h = blockIdx.y;
  const int s0   = blockIdx.x * 128;
  const int wave = tid >> 5, lane = tid & 31;
  const int lm = lane & 15, lh = lane >> 4;
  const int rowBase = (wave & 3) * 32;
  const int colBase = (wave >> 2) * 32;

  const float* Xb = X + (size_t)b * CIN * SEQ;
  const float* Wh = W + (size_t)h * CIN * DH;

  v8f acc[2][2] = {};
  for (int kc = 0; kc < CIN; kc += 32) {
    for (int i = tid; i < 128 * 32; i += 256) {
      int c = i >> 7, s = i & 127;               // coalesced along s
      lA[s][c] = (_Float16)Xb[(size_t)(kc + c) * SEQ + s0 + s];
    }
    for (int i = tid; i < 32 * 64; i += 256) {
      int c = i >> 6, d = i & 63;                // coalesced along d
      lBt[d][c] = (_Float16)Wh[(size_t)(kc + c) * DH + d];
    }
    __syncthreads();
    v16h a[2], bb[2];
#pragma unroll
    for (int t = 0; t < 2; ++t)
#pragma unroll
      for (int e = 0; e < 16; ++e)
        a[t][e] = lA[rowBase + t * 16 + lm][a_kidx(e, lh)];
#pragma unroll
    for (int t = 0; t < 2; ++t)
#pragma unroll
      for (int e = 0; e < 16; ++e)
        bb[t][e] = lBt[colBase + t * 16 + lm][b_kidx(e, lh)];
#pragma unroll
    for (int tm = 0; tm < 2; ++tm)
#pragma unroll
      for (int tn = 0; tn < 2; ++tn)
        acc[tm][tn] = wmma16(a[tm], bb[tn], acc[tm][tn]);
    __syncthreads();
  }
  // C/D layout: VGPR r -> row r + 8*half, col = lm
#pragma unroll
  for (int tm = 0; tm < 2; ++tm)
#pragma unroll
    for (int tn = 0; tn < 2; ++tn)
#pragma unroll
      for (int r = 0; r < 8; ++r) {
        int row = s0 + rowBase + tm * 16 + r + lh * 8;
        int col = colBase + tn * 16 + lm;
        float v = acc[tm][tn][r] + bias[h * DH + col];
        out[(((size_t)b * NH + h) * SEQ + row) * DH + col] = (_Float16)v;
      }
}

// ---------------------------------------------------------------------------
// Kernel 2: flash attention per (b,h).  Block = 128 query rows, 8 waves,
// wave owns 16 rows (Q frags resident in regs).  T-chunks of 64 keys.
// K tile staged row-major (async path); V tile staged TRANSPOSED (lVt[d][t]).
// ---------------------------------------------------------------------------
__global__ __launch_bounds__(256)
void attn_kernel(const _Float16* __restrict__ Q, const _Float16* __restrict__ K,
                 const _Float16* __restrict__ V, _Float16* __restrict__ O) {
  __shared__ alignas(16) _Float16 lK[64][72];    // [t][d], 144B row
  __shared__ alignas(16) _Float16 lVt[64][72];   // [d][t], transposed
  __shared__ alignas(16) _Float16 lP[8][16][72]; // per-wave P tile [row][t]
  const int tid  = threadIdx.x;
  const int b    = blockIdx.z, h = blockIdx.y;
  const int s0   = blockIdx.x * 128;
  const int wave = tid >> 5, lane = tid & 31;
  const int lm = lane & 15, lh = lane >> 4;
  const size_t bh = (size_t)b * NH + h;
  const _Float16* Qb = Q + bh * SEQ * DH;
  const _Float16* Kb = K + bh * SEQ * DH;
  const _Float16* Vb = V + bh * SEQ * DH;
  const int sRow = s0 + wave * 16;

  // Q A-fragments for d-chunks [0,32) and [32,64)
  v16h qf[2];
#pragma unroll
  for (int c = 0; c < 2; ++c)
#pragma unroll
    for (int e = 0; e < 16; ++e)
      qf[c][e] = Qb[(size_t)(sRow + lm) * DH + c * 32 + a_kidx(e, lh)];

  float mrow[8], lrow[8];
#pragma unroll
  for (int r = 0; r < 8; ++r) { mrow[r] = -1e30f; lrow[r] = 0.f; }
  v8f oacc[4] = {};

  for (int t0 = 0; t0 < SEQ; t0 += 64) {
#if USE_ASYNC
    // K tile: contiguous 64 rows x 128B -> async DMA, 16B per op, 2 ops/thread
    for (int i = tid; i < 512; i += 256) {
      int t = i >> 3, ch = i & 7;
      __builtin_amdgcn_global_load_async_to_lds_b128(
          (v4i*)(Kb + (size_t)(t0 + t) * DH + ch * 8),
          (v4i*)(&lK[t][ch * 8]),
          0, 0);
    }
#else
    for (int i = tid; i < 512; i += 256) {
      int t = i >> 3, ch = i & 7;
      *(v8h*)&lK[t][ch * 8] = *(const v8h*)(Kb + (size_t)(t0 + t) * DH + ch * 8);
    }
#endif
    for (int i = tid; i < 64 * 64; i += 256) {
      int t = i >> 6, d = i & 63;                  // coalesced read, transpose
      lVt[d][t] = Vb[(size_t)(t0 + t) * DH + d];
    }
    if (t0 + 64 < SEQ) {  // global_prefetch_b8 of next chunk
      __builtin_prefetch(Kb + (size_t)(t0 + 64 + (tid >> 2)) * DH, 0, 1);
      __builtin_prefetch(Vb + (size_t)(t0 + 64 + (tid >> 2)) * DH, 0, 1);
    }
#if USE_ASYNC
    __builtin_amdgcn_s_wait_asynccnt(0);
#endif
    __syncthreads();

    // Scores S = (Q Kt) * 1/8 ; 4 n-tiles of 16 keys, 2 k-chunks of d
    float sc[4][8];
#pragma unroll
    for (int tn = 0; tn < 4; ++tn) {
      v8f sa = {};
#pragma unroll
      for (int c = 0; c < 2; ++c) {
        v16h bf;
#pragma unroll
        for (int e = 0; e < 16; ++e)
          bf[e] = lK[tn * 16 + lm][c * 32 + b_kidx(e, lh)]; // contiguous
        sa = wmma16(qf[c], bf, sa);
      }
#pragma unroll
      for (int r = 0; r < 8; ++r) sc[tn][r] = sa[r] * 0.125f;
    }

    // Online softmax: row-reductions across the 16-lane group (C/D layout)
    float corr[8], rs[8];
#pragma unroll
    for (int r = 0; r < 8; ++r) {
      float m = sc[0][r];
#pragma unroll
      for (int tn = 1; tn < 4; ++tn) m = fmaxf(m, sc[tn][r]);
      for (int off = 8; off > 0; off >>= 1)
        m = fmaxf(m, __shfl_xor(m, off, 16));
      float mnew = fmaxf(mrow[r], m);
      corr[r] = __expf(mrow[r] - mnew);
      mrow[r] = mnew;
      rs[r] = 0.f;
    }
#pragma unroll
    for (int tn = 0; tn < 4; ++tn)
#pragma unroll
      for (int r = 0; r < 8; ++r) {
        float p = __expf(sc[tn][r] - mrow[r]);
        rs[r] += p;
        lP[wave][r + lh * 8][tn * 16 + lm] = (_Float16)p;
      }
#pragma unroll
    for (int r = 0; r < 8; ++r) {
      float t = rs[r];
      for (int off = 8; off > 0; off >>= 1)
        t += __shfl_xor(t, off, 16);
      lrow[r] = lrow[r] * corr[r] + t;
#pragma unroll
      for (int dn = 0; dn < 4; ++dn) oacc[dn][r] *= corr[r];
    }
    __syncthreads();

    // O += P * V ; P A-frags from LDS, V B-frags from transposed tile
    v16h pf[2];
#pragma unroll
    for (int c = 0; c < 2; ++c)
#pragma unroll
      for (int e = 0; e < 16; ++e)
        pf[c][e] = lP[wave][lm][c * 32 + a_kidx(e, lh)];
#pragma unroll
    for (int dn = 0; dn < 4; ++dn)
#pragma unroll
      for (int c = 0; c < 2; ++c) {
        v16h vf;
#pragma unroll
        for (int e = 0; e < 16; ++e)
          vf[e] = lVt[dn * 16 + lm][c * 32 + b_kidx(e, lh)]; // contiguous
        oacc[dn] = wmma16(pf[c], vf, oacc[dn]);
      }
    __syncthreads();
  }

  // Normalize and write concat-head layout [B][S][H*d] (f16)
#pragma unroll
  for (int r = 0; r < 8; ++r) {
    float inv = 1.0f / lrow[r];
    int row = sRow + r + lh * 8;
#pragma unroll
    for (int dn = 0; dn < 4; ++dn) {
      int d = dn * 16 + lm;
      O[((size_t)b * SEQ + row) * (NH * DH) + h * DH + d] =
          (_Float16)(oacc[dn][r] * inv);
    }
  }
}

// ---------------------------------------------------------------------------
// Kernel 3: output projection.  Y = A(8192x768,f16) * Wp(768x768) + bp  (f32)
// ---------------------------------------------------------------------------
__global__ __launch_bounds__(256)
void outproj_kernel(const _Float16* __restrict__ A, const float* __restrict__ Wp,
                    const float* __restrict__ bp, float* __restrict__ Y) {
  __shared__ alignas(16) _Float16 lA[128][40];
  __shared__ alignas(16) _Float16 lBt[64][40];  // [n][c], transposed
  const int tid = threadIdx.x;
  const int m0 = blockIdx.x * 128;
  const int n0 = blockIdx.y * 64;
  const int wave = tid >> 5, lane = tid & 31;
  const int lm = lane & 15, lh = lane >> 4;
  const int rowBase = (wave & 3) * 32;
  const int colBase = (wave >> 2) * 32;
  const int N = NH * DH; // 768

  v8f acc[2][2] = {};
  for (int kc = 0; kc < N; kc += 32) {
    for (int i = tid; i < 512; i += 256) {       // 128 rows x 4 chunks of 16B
      int s = i >> 2, ch = i & 3;
      *(v8h*)&lA[s][ch * 8] =
          *(const v8h*)(A + (size_t)(m0 + s) * N + kc + ch * 8);
    }
    for (int i = tid; i < 32 * 64; i += 256) {
      int c = i >> 6, n = i & 63;                // coalesced along n
      lBt[n][c] = (_Float16)Wp[(size_t)(kc + c) * N + n0 + n];
    }
    __syncthreads();
    v16h a[2], bb[2];
#pragma unroll
    for (int t = 0; t < 2; ++t)
#pragma unroll
      for (int e = 0; e < 16; ++e)
        a[t][e] = lA[rowBase + t * 16 + lm][a_kidx(e, lh)];
#pragma unroll
    for (int t = 0; t < 2; ++t)
#pragma unroll
      for (int e = 0; e < 16; ++e)
        bb[t][e] = lBt[colBase + t * 16 + lm][b_kidx(e, lh)];
#pragma unroll
    for (int tm = 0; tm < 2; ++tm)
#pragma unroll
      for (int tn = 0; tn < 2; ++tn)
        acc[tm][tn] = wmma16(a[tm], bb[tn], acc[tm][tn]);
    __syncthreads();
  }
#pragma unroll
  for (int tm = 0; tm < 2; ++tm)
#pragma unroll
    for (int tn = 0; tn < 2; ++tn)
#pragma unroll
      for (int r = 0; r < 8; ++r) {
        int row = m0 + rowBase + tm * 16 + r + lh * 8;
        int col = n0 + colBase + tn * 16 + lm;
        Y[(size_t)row * N + col] = acc[tm][tn][r] + bp[col];
      }
}

// ---------------------------------------------------------------------------
extern "C" void kernel_launch(void* const* d_in, const int* in_sizes, int n_in,
                              void* d_out, int out_size, void* d_ws, size_t ws_size,
                              hipStream_t stream) {
  (void)in_sizes; (void)n_in; (void)out_size; (void)ws_size;
  const float* x  = (const float*)d_in[0];
  const float* xt = (const float*)d_in[1];
  const float* Wq = (const float*)d_in[2];
  const float* bq = (const float*)d_in[3];
  const float* Wk = (const float*)d_in[4];
  const float* bk = (const float*)d_in[5];
  const float* Wv = (const float*)d_in[6];
  const float* bv = (const float*)d_in[7];
  const float* Wp = (const float*)d_in[8];
  const float* bp = (const float*)d_in[9];
  float* out = (float*)d_out;

  // Workspace: Q, K, V [B,H,S,64] f16 + attn-out [B,S,768] f16  (~48 MB)
  const size_t qElems = (size_t)B_SZ * NH * SEQ * DH;
  _Float16* Qw = (_Float16*)d_ws;
  _Float16* Kw = Qw + qElems;
  _Float16* Vw = Kw + qElems;
  _Float16* Aw = Vw + qElems;

  dim3 blk(256);
  dim3 gproj(SEQ / 128, NH, B_SZ);   // 8 x 12 x 8 = 768 blocks
  proj_kernel<<<gproj, blk, 0, stream>>>(x,  Wq, bq, Qw);
  proj_kernel<<<gproj, blk, 0, stream>>>(xt, Wk, bk, Kw);
  proj_kernel<<<gproj, blk, 0, stream>>>(xt, Wv, bv, Vw);
  attn_kernel<<<gproj, blk, 0, stream>>>(Qw, Kw, Vw, Aw);
  dim3 gout((B_SZ * SEQ) / 128, (NH * DH) / 64); // 64 x 12
  outproj_kernel<<<gout, blk, 0, stream>>>(Aw, Wp, bp, out);
}
